// GraphMoEBackbone_55834574848435
// MI455X (gfx1250) — matrix operand order repaired and checked by reading, not compiled, not run on hardware
//
#include <hip/hip_runtime.h>

// ---------------- problem constants (from reference) ----------------
#define NN   20000
#define EE   160000
#define GG   64
#define Hdim 512
#define RHd  128
#define NEx  4
#define NLay 3

typedef __attribute__((ext_vector_type(16))) __bf16 v16bf;
typedef __attribute__((ext_vector_type(8)))  __bf16 v8bf;
typedef __attribute__((ext_vector_type(8)))  float  v8f;

__device__ inline v16bf mk_afrag(const __bf16* p) {
  // A 16x32 bf16 fragment: this lane holds two contiguous 8-element (16B) chunks
  v8bf lo = *(const v8bf*)p;
  v8bf hi = *(const v8bf*)(p + 8);
  v16bf r;
#pragma unroll
  for (int i = 0; i < 8; i++) { r[i] = lo[i]; r[i + 8] = hi[i]; }
  return r;
}

// ---------------- WMMA GEMM: C[M,512] = A1@B1t (+ A2@B2t) + bias ----------------
// One block = one 16-row strip of A, async-staged into LDS once (no per-step barriers).
// Each of the 8 waves owns a 64-col strip: A fragments from LDS (ds_load), B from
// global (L2-resident, double-buffered). Two accumulator banks -> 8 WMMA chains.
// B*t are bf16 column-major: Bt[n*512 + k] (a 32x16 B fragment is one 32B lane load).
// MODE 0: ReLU -> bf16 store       (expert layers 0,1)
// MODE 1: d_out += gate[:,e]*val   (expert layer 2, fp32 accumulate)
// MODE 2: plain bias -> bf16       (encoder layer 2)
template <int MODE, bool DUAL>
__global__ __launch_bounds__(256) void gemm_bf16_wmma(
    const __bf16* __restrict__ A1, const __bf16* __restrict__ B1t,
    const __bf16* __restrict__ A2, const __bf16* __restrict__ B2t,
    const float*  __restrict__ bias, int M,
    __bf16* __restrict__ outbf, float* __restrict__ outf,
    const float* __restrict__ gate, int expert)
{
  __shared__ __bf16 lA[(DUAL ? 2 : 1) * 16 * Hdim];   // 16KB (32KB dual)

  const int lane = threadIdx.x & 31;
  const int wave = threadIdx.x >> 5;
  const int row0 = blockIdx.x * 16;       // block's 16-row strip
  const int col0 = wave * 64;             // wave's 64-col strip (4 x 16-col tiles)
  if (row0 >= M) return;

  // ---- async-stage this block's A rows into LDS (ASYNCcnt path) ----
  {
    const unsigned lbase = (unsigned)(size_t)(&lA[0]);
    const unsigned long long g1 =
        (unsigned long long)(size_t)(A1 + (size_t)row0 * Hdim);
#pragma unroll
    for (int i = 0; i < 4; i++) {                     // 16KB = 1024 x 16B chunks
      const int boff = ((int)threadIdx.x + 256 * i) * 16;
      unsigned l = lbase + boff;
      unsigned long long g = g1 + boff;
      asm volatile("global_load_async_to_lds_b128 %0, %1, off"
                   :: "v"(l), "v"(g) : "memory");
    }
    if constexpr (DUAL) {
      const unsigned long long g2 =
          (unsigned long long)(size_t)(A2 + (size_t)row0 * Hdim);
#pragma unroll
      for (int i = 0; i < 4; i++) {
        const int boff = ((int)threadIdx.x + 256 * i) * 16;
        unsigned l = lbase + 16 * Hdim * 2 + boff;    // second matrix at +16KB
        unsigned long long g = g2 + boff;
        asm volatile("global_load_async_to_lds_b128 %0, %1, off"
                     :: "v"(l), "v"(g) : "memory");
      }
    }
    asm volatile("s_wait_asynccnt 0x0" ::: "memory");
    __syncthreads();
  }

  // lane mappings: contiguous k order (k = 16*(lane/16) + s), same for A and B
  const int am  = lane & 15;                // A row within tile
  const int akb = (lane & 16) ? 16 : 0;     // A k-chunk base
  const int bn  = lane & 15;                // B col within tile
  const int bks = (lane & 16) ? 16 : 0;     // B k-chunk base

  const __bf16* la1 = lA + (size_t)am * Hdim + akb;            // LDS-resident A
  const __bf16* la2 = DUAL ? (la1 + 16 * Hdim) : la1;
  const __bf16* b1p = B1t + (size_t)(col0 + bn) * Hdim + bks;
  const __bf16* b2p = DUAL ? (B2t + (size_t)(col0 + bn) * Hdim + bks) : b1p;

  v8f accR[4] = {};
  v8f accO[4] = {};

  // prologue: B fragments for k0 = 0
  v16bf b1c[4];
#pragma unroll
  for (int j = 0; j < 4; j++) b1c[j] = *(const v16bf*)(b1p + (size_t)j * 16 * Hdim);
  v16bf b2c[4];
  if constexpr (DUAL) {
#pragma unroll
    for (int j = 0; j < 4; j++) b2c[j] = *(const v16bf*)(b2p + (size_t)j * 16 * Hdim);
  }

#pragma unroll
  for (int k0 = 0; k0 < Hdim; k0 += 32) {
    // issue next k-step's B loads first (wraps on last iter; harmless)
    const int kn = (k0 + 32 < Hdim) ? (k0 + 32) : 0;
    v16bf b1n[4];
#pragma unroll
    for (int j = 0; j < 4; j++) b1n[j] = *(const v16bf*)(b1p + (size_t)j * 16 * Hdim + kn);
    v16bf b2n[4];
    if constexpr (DUAL) {
#pragma unroll
      for (int j = 0; j < 4; j++) b2n[j] = *(const v16bf*)(b2p + (size_t)j * 16 * Hdim + kn);
    }

    // A fragments from LDS (ds_load_b128 x2 per fragment)
    const v16bf a1 = mk_afrag(la1 + k0);
#pragma unroll
    for (int j = 0; j < 4; j++)
      accR[j] = __builtin_amdgcn_wmma_f32_16x16x32_bf16(false, a1, false, b1c[j],
                                                        (short)0, accR[j], false, false);
    if constexpr (DUAL) {
      const v16bf a2 = mk_afrag(la2 + k0);
#pragma unroll
      for (int j = 0; j < 4; j++)
        accO[j] = __builtin_amdgcn_wmma_f32_16x16x32_bf16(false, a2, false, b2c[j],
                                                          (short)0, accO[j], false, false);
    }

    // rotate (pure renaming after full unroll)
#pragma unroll
    for (int j = 0; j < 4; j++) b1c[j] = b1n[j];
    if constexpr (DUAL) {
#pragma unroll
      for (int j = 0; j < 4; j++) b2c[j] = b2n[j];
    }
  }

  // C/D layout: VGPR r -> row = row0 + r + (lane>=16 ? 8 : 0), col = col0 + 16*j + (lane&15)
  const int rbase = row0 + ((lane >> 4) << 3);
#pragma unroll
  for (int r = 0; r < 8; r++) {
    const int row = rbase + r;
    float g = 0.0f;
    if (MODE == 1) g = gate[(size_t)row * NEx + expert];
#pragma unroll
    for (int j = 0; j < 4; j++) {
      const int colg = col0 + j * 16 + (lane & 15);
      float v = accR[j][r] + (bias ? bias[colg] : 0.0f);
      if constexpr (DUAL) v += accO[j][r];
      if (MODE == 0) {
        outbf[(size_t)row * Hdim + colg] = (__bf16)fmaxf(v, 0.0f);
      } else if (MODE == 2) {
        outbf[(size_t)row * Hdim + colg] = (__bf16)v;
      } else {
        outf[(size_t)row * Hdim + colg] += g * v;
      }
    }
  }
}

// ---------------- weight prep: [k][n] f32 row-major -> [n][k] bf16 ----------------
__global__ void prep_wt_k(const float* __restrict__ W, __bf16* __restrict__ Wt, int nmat) {
  long idx = (long)blockIdx.x * blockDim.x + threadIdx.x;
  if (idx >= (long)nmat * Hdim * Hdim) return;
  int m = (int)(idx >> 18);
  int rem = (int)(idx & ((1 << 18) - 1));
  int k = rem >> 9, n = rem & 511;
  Wt[(size_t)m * Hdim * Hdim + (size_t)n * Hdim + k] = (__bf16)W[idx];
}

// ---------------- encoder layer 1: relu(x[:,4:10] @ W1 + b1) -> bf16 ----------------
__global__ void encoder1_k(const float* __restrict__ x, const float* __restrict__ W1,
                           const float* __restrict__ b1, __bf16* __restrict__ out) {
  int idx = blockIdx.x * blockDim.x + threadIdx.x;
  if (idx >= NN * Hdim) return;
  int i = idx >> 9, j = idx & 511;
  const float* xr = x + (size_t)i * 16 + 4;
  float acc = b1[j];
#pragma unroll
  for (int k = 0; k < 6; k++) acc += xr[k] * W1[k * Hdim + j];
  out[idx] = (__bf16)fmaxf(acc, 0.0f);
}

// ---------------- per-graph counts ----------------
__global__ void count_nodes_k(const int* __restrict__ batch, float* __restrict__ nodes) {
  int i = blockIdx.x * blockDim.x + threadIdx.x;
  if (i < NN) atomicAdd(&nodes[batch[i]], 1.0f);
}
__global__ void count_edges_k(const int* __restrict__ src, const int* __restrict__ batch,
                              float* __restrict__ edges) {
  int i = blockIdx.x * blockDim.x + threadIdx.x;
  if (i < EE) atomicAdd(&edges[batch[src[i]]], 1.0f);
}

// ---------------- graph size features (single block of G threads) ----------------
__global__ void graph_stats_k(const float* __restrict__ nodes, const float* __restrict__ edges,
                              float* __restrict__ fstat, float* __restrict__ size_norm) {
  __shared__ float lnv[GG], lev[GG], st[6];
  int g = threadIdx.x;
  float ln = logf(fmaxf(nodes[g], 1.0f));
  float le = log1pf(edges[g]);
  lnv[g] = ln; lev[g] = le;
  __syncthreads();
  if (g == 0) {
    float s1 = 0, s2 = 0, t1 = 0, t2 = 0, mn = lnv[0], mx = lnv[0];
    for (int i = 0; i < GG; i++) {
      s1 += lnv[i]; s2 += lnv[i] * lnv[i];
      t1 += lev[i]; t2 += lev[i] * lev[i];
      mn = fminf(mn, lnv[i]); mx = fmaxf(mx, lnv[i]);
    }
    float m1 = s1 / GG, m2 = t1 / GG;
    st[0] = m1; st[1] = sqrtf(fmaxf(s2 / GG - m1 * m1, 0.0f));
    st[2] = m2; st[3] = sqrtf(fmaxf(t2 / GG - m2 * m2, 0.0f));
    st[4] = mn; st[5] = mx;
  }
  __syncthreads();
  fstat[g * 2 + 0] = (ln - st[0]) / (st[1] + 1e-6f);
  fstat[g * 2 + 1] = (le - st[2]) / (st[3] + 1e-6f);
  size_norm[g]     = (ln - st[4]) / (st[5] - st[4] + 1e-6f);
}

// ---------------- fused router: Linear->LN->ReLU->Linear + prior -> softmax -> top2 gate ----
__global__ __launch_bounds__(RHd) void router_k(
    const __bf16* __restrict__ hbf, const float* __restrict__ fstat,
    const float* __restrict__ size_norm, const int* __restrict__ batch,
    const float* __restrict__ W1, const float* __restrict__ b1,
    const float* __restrict__ lng, const float* __restrict__ lnb,
    const float* __restrict__ W2, const float* __restrict__ b2,
    const float* __restrict__ centers, float* __restrict__ gate)
{
  __shared__ float sh[Hdim];
  __shared__ float sr[RHd];
  __shared__ float sred[RHd];
  __shared__ float sl[NEx];
  const int i = blockIdx.x;
  const int t = threadIdx.x;
  const __bf16* hr = hbf + (size_t)i * Hdim;
#pragma unroll
  for (int u = 0; u < 4; u++) sh[t * 4 + u] = (float)hr[t * 4 + u];
  __syncthreads();
  const int g = batch[i];
  const float f0 = fstat[g * 2 + 0], f1 = fstat[g * 2 + 1];
  float acc = b1[t] + f0 * W1[512 * RHd + t] + f1 * W1[513 * RHd + t];
  for (int k = 0; k < Hdim; k++) acc += sh[k] * W1[k * RHd + t];
  // LayerNorm over 128
  sred[t] = acc; __syncthreads();
  for (int s = RHd / 2; s > 0; s >>= 1) { if (t < s) sred[t] += sred[t + s]; __syncthreads(); }
  const float mean = sred[0] / (float)RHd;
  __syncthreads();
  const float d = acc - mean;
  sred[t] = d * d; __syncthreads();
  for (int s = RHd / 2; s > 0; s >>= 1) { if (t < s) sred[t] += sred[t + s]; __syncthreads(); }
  const float var = sred[0] / (float)RHd;
  sr[t] = fmaxf(d * rsqrtf(var + 1e-5f) * lng[t] + lnb[t], 0.0f);
  __syncthreads();
  if (t < NEx) {
    float l = b2[t];
    for (int k = 0; k < RHd; k++) l += sr[k] * W2[k * NEx + t];
    const float dq = size_norm[g] - centers[t];
    sl[t] = 0.7f * l + 0.3f * (-dq * dq);   // SPS=0.3, STEMP=1.0
  }
  __syncthreads();
  if (t == 0) {
    float m = fmaxf(fmaxf(sl[0], sl[1]), fmaxf(sl[2], sl[3]));
    float p[NEx], s = 0.0f;
    for (int j = 0; j < NEx; j++) { p[j] = __expf(sl[j] - m); s += p[j]; }
    for (int j = 0; j < NEx; j++) p[j] /= s;
    int i1 = 0; for (int j = 1; j < NEx; j++) if (p[j] > p[i1]) i1 = j;
    int i2 = -1; for (int j = 0; j < NEx; j++) { if (j == i1) continue; if (i2 < 0 || p[j] > p[i2]) i2 = j; }
    const float den = p[i1] + p[i2] + 1e-8f;
    float w[NEx] = {0, 0, 0, 0};
    w[i1] = p[i1] / den; w[i2] = p[i2] / den;
    for (int j = 0; j < NEx; j++) gate[(size_t)i * NEx + j] = w[j];
  }
}

// ---------------- edge scatter-add (f32 atomics, L2-resident) ----------------
__global__ void scatter_k(const __bf16* __restrict__ z, const int* __restrict__ src,
                          const int* __restrict__ dst, float* __restrict__ agg) {
  long idx = (long)blockIdx.x * blockDim.x + threadIdx.x;
  if (idx >= (long)EE * 128) return;
  int e = (int)(idx >> 7);
  int f = ((int)idx & 127) << 2;
  const int s = src[e], d = dst[e];
  const __bf16* zp = z + (size_t)s * Hdim + f;
  float* ap = agg + (size_t)d * Hdim + f;
  atomicAdd(ap + 0, (float)zp[0]);
  atomicAdd(ap + 1, (float)zp[1]);
  atomicAdd(ap + 2, (float)zp[2]);
  atomicAdd(ap + 3, (float)zp[3]);
}

__global__ void cvt_k(const float* __restrict__ a, __bf16* __restrict__ b, long n) {
  long i = (long)blockIdx.x * blockDim.x + threadIdx.x;
  if (i < n) b[i] = (__bf16)a[i];
}

// ---------------- launch ----------------
extern "C" void kernel_launch(void* const* d_in, const int* in_sizes, int n_in,
                              void* d_out, int out_size, void* d_ws, size_t ws_size,
                              hipStream_t stream) {
  (void)in_sizes; (void)n_in; (void)ws_size;
  const float* x     = (const float*)d_in[0];
  const int*   ei    = (const int*)d_in[1];
  const int*   batch = (const int*)d_in[2];
  const float* encW1 = (const float*)d_in[4];
  const float* encb1 = (const float*)d_in[5];
  const float* encW2 = (const float*)d_in[6];
  const float* encb2 = (const float*)d_in[7];
  const float* rW1   = (const float*)d_in[8];
  const float* rb1   = (const float*)d_in[9];
  const float* lng   = (const float*)d_in[10];
  const float* lnb   = (const float*)d_in[11];
  const float* rW2   = (const float*)d_in[12];
  const float* rb2   = (const float*)d_in[13];
  const float* ctr   = (const float*)d_in[14];
  const float* Wrel  = (const float*)d_in[15];
  const float* brel  = (const float*)d_in[16];
  const float* Wroot = (const float*)d_in[17];
  const int* src = ei;
  const int* dst = ei + EE;

  // scratch carve-out
  char* ws = (char*)d_ws;
  size_t off = 0;
  auto carve = [&](size_t bytes) -> char* {
    char* p = ws + off;
    off = (off + bytes + 255) & ~(size_t)255;
    return p;
  };
  float*  agg   = (float*) carve((size_t)NN * Hdim * 4);
  __bf16* hbf   = (__bf16*)carve((size_t)NN * Hdim * 2);
  __bf16* zbf   = (__bf16*)carve((size_t)NN * Hdim * 2);
  __bf16* z2bf  = (__bf16*)carve((size_t)NN * Hdim * 2);   // also aliases encoder intermediate
  __bf16* aggbf = (__bf16*)carve((size_t)NN * Hdim * 2);
  __bf16* encW2t= (__bf16*)carve((size_t)Hdim * Hdim * 2);
  __bf16* Wrelt = (__bf16*)carve((size_t)NEx * NLay * Hdim * Hdim * 2);
  __bf16* Wroott= (__bf16*)carve((size_t)NEx * NLay * Hdim * Hdim * 2);
  float*  gate  = (float*) carve((size_t)NN * NEx * 4);
  float*  nodes = (float*) carve(GG * 4);
  float*  edgec = (float*) carve(GG * 4);
  float*  fstat = (float*) carve(GG * 2 * 4);
  float*  sznrm = (float*) carve(GG * 4);

  // one 16-row strip per block; 8 waves cover the 8 x 64-col strips
  const int gemm_blocks = NN / 16;   // 1250

  hipMemsetAsync(d_out, 0, (size_t)out_size * sizeof(float), stream);
  hipMemsetAsync(nodes, 0, GG * 4, stream);
  hipMemsetAsync(edgec, 0, GG * 4, stream);

  // weight prep (f32 row-major -> bf16 col-major)
  prep_wt_k<<<(1 * Hdim * Hdim + 255) / 256, 256, 0, stream>>>(encW2, encW2t, 1);
  prep_wt_k<<<(12 * Hdim * Hdim + 255) / 256, 256, 0, stream>>>(Wrel, Wrelt, 12);
  prep_wt_k<<<(12 * Hdim * Hdim + 255) / 256, 256, 0, stream>>>(Wroot, Wroott, 12);

  // encoder
  __bf16* xenc = z2bf;  // alias: free until expert layer 1
  encoder1_k<<<(NN * Hdim + 255) / 256, 256, 0, stream>>>(x, encW1, encb1, xenc);
  gemm_bf16_wmma<2, false><<<gemm_blocks, 256, 0, stream>>>(
      xenc, encW2t, nullptr, nullptr, encb2, NN, hbf, nullptr, gate, 0);

  // graph size stats + router
  count_nodes_k<<<(NN + 255) / 256, 256, 0, stream>>>(batch, nodes);
  count_edges_k<<<(EE + 255) / 256, 256, 0, stream>>>(src, batch, edgec);
  graph_stats_k<<<1, GG, 0, stream>>>(nodes, edgec, fstat, sznrm);
  router_k<<<NN, RHd, 0, stream>>>(hbf, fstat, sznrm, batch, rW1, rb1, lng, lnb,
                                   rW2, rb2, ctr, gate);

  // experts: 3 GraphConv layers each
  const long nelem = (long)NN * Hdim;
  for (int e = 0; e < NEx; e++) {
    for (int l = 0; l < NLay; l++) {
      const __bf16* zin  = (l == 0) ? hbf : ((l == 1) ? zbf : z2bf);
      __bf16*       zout = (l == 0) ? zbf : z2bf;
      const int mi = e * NLay + l;
      hipMemsetAsync(agg, 0, (size_t)NN * Hdim * 4, stream);
      scatter_k<<<(int)(((long)EE * 128 + 255) / 256), 256, 0, stream>>>(zin, src, dst, agg);
      cvt_k<<<(int)((nelem + 255) / 256), 256, 0, stream>>>(agg, aggbf, nelem);
      if (l == 2) {
        gemm_bf16_wmma<1, true><<<gemm_blocks, 256, 0, stream>>>(
            aggbf, Wrelt + (size_t)mi * Hdim * Hdim,
            zin,   Wroott + (size_t)mi * Hdim * Hdim,
            brel + (size_t)mi * Hdim, NN, zout, (float*)d_out, gate, e);
      } else {
        gemm_bf16_wmma<0, true><<<gemm_blocks, 256, 0, stream>>>(
            aggbf, Wrelt + (size_t)mi * Hdim * Hdim,
            zin,   Wroott + (size_t)mi * Hdim * Hdim,
            brel + (size_t)mi * Hdim, NN, zout, (float*)d_out, gate, e);
      }
    }
  }
}
